// MultiHeadAttention_84628035600701
// MI455X (gfx1250) — matrix operand order repaired and checked by reading, not compiled, not run on hardware
//
#include <hip/hip_runtime.h>

#define B_ 8
#define L_ 1024
#define D_ 1024
#define H_ 16
#define DK_ 64

typedef __attribute__((ext_vector_type(16))) __bf16        v16bf;
typedef __attribute__((ext_vector_type(8)))  float         v8f;
typedef __attribute__((ext_vector_type(4)))  unsigned int  v4u;

union BF16Frag { v16bf v; v4u q[2]; unsigned short us[16]; };

__device__ __forceinline__ unsigned short f2bf(float f) {
  unsigned int u = __float_as_uint(f);
  u += 0x7FFFu + ((u >> 16) & 1u);   // round-to-nearest-even
  return (unsigned short)(u >> 16);
}

__device__ __forceinline__ v8f wmma_bf16(v16bf a, v16bf b, v8f c) {
  return __builtin_amdgcn_wmma_f32_16x16x32_bf16(false, a, false, b, (short)0, c, false, false);
}

// A fragment (16x32 bf16): lane<16 holds K = kk+{0..15},
// lane>=16 holds K = kk+{16..31} of row M = lane&15.
__device__ __forceinline__ v16bf frag_a_bf16(const unsigned short* row, int kk, int lane) {
  const int sel = (lane >> 4) & 1;
  BF16Frag f;
  f.q[0] = *(const v4u*)(row + kk + sel * 16);
  f.q[1] = *(const v4u*)(row + kk + sel * 16 + 8);
  return f.v;
}

__device__ __forceinline__ v16bf frag_a_f32(const float* row, int kk, int lane) {
  const int sel = (lane >> 4) & 1;
  BF16Frag f;
#pragma unroll
  for (int j = 0; j < 8; ++j) f.us[j]     = f2bf(row[kk + sel * 16 + j]);
#pragma unroll
  for (int j = 0; j < 8; ++j) f.us[8 + j] = f2bf(row[kk + sel * 16 + 8 + j]);
  return f.v;
}

// B fragment (32x16 bf16): lane holds 16 consecutive K of column n = lane&15,
// K-base = (lane>>4)*16.  p must point at that K-run (column-contiguous store).
__device__ __forceinline__ v16bf frag_b_bf16(const unsigned short* p) {
  BF16Frag f;
  f.q[0] = *(const v4u*)(p);
  f.q[1] = *(const v4u*)(p + 8);
  return f.v;
}

// ---------------------------------------------------------------------------
// Kernel 0: proj_w (1024x1024 f32, row n = B-column) -> bf16, same layout.
// ---------------------------------------------------------------------------
__global__ __launch_bounds__(256) void k_convw(const float* __restrict__ pw,
                                               unsigned short* __restrict__ pwb)
{
  int i = blockIdx.x * 1024 + threadIdx.x * 4;
  float4 f = *(const float4*)(pw + i);
  pwb[i + 0] = f2bf(f.x); pwb[i + 1] = f2bf(f.y);
  pwb[i + 2] = f2bf(f.z); pwb[i + 3] = f2bf(f.w);
}

// ---------------------------------------------------------------------------
// Kernel 1: per-head QKV projections.  X(1024x1024) @ W_h(1024x64) for all h.
// X tile staged once in LDS as bf16; W K-slab staged transposed per K-step.
// Q outputs pre-scaled by 1/sqrt(d_model)=1/32.  V stored transposed [h][b][64][L].
// ---------------------------------------------------------------------------
__global__ __launch_bounds__(256) void k_qkv_proj(
    const float* __restrict__ qin, const float* __restrict__ kin, const float* __restrict__ vin,
    const float* __restrict__ wq, const float* __restrict__ wk, const float* __restrict__ wv,
    unsigned short* __restrict__ Qs, unsigned short* __restrict__ Ks,
    unsigned short* __restrict__ VsT)
{
  extern __shared__ char smem[];
  unsigned short* Xt = (unsigned short*)smem;   // [32][1024] bf16 (64 KB)
  unsigned short* Wl = Xt + 32 * 1024;          // [64][32] bf16, column-contig K (4 KB)

  const int mat  = blockIdx.x >> 5;             // 0=q,1=k,2=v  (gridDim.x = 96)
  const int row0 = (blockIdx.x & 31) * 32;
  const int b    = blockIdx.y;
  const int tid  = threadIdx.x;
  const int lane = tid & 31;
  const int wave = tid >> 5;
  const int rt   = wave >> 2;                   // row sub-tile 0..1
  const int nt   = wave & 3;                    // col sub-tile 0..3

  const float* X = (mat == 0) ? qin : (mat == 1) ? kin : vin;
  const float* W = (mat == 0) ? wq  : (mat == 1) ? wk  : wv;
  const float scale = (mat == 0) ? 0.03125f : 1.0f;   // fold 1/sqrt(D_MODEL) into Q

  const float4* src = (const float4*)(X + (size_t)(b * L_ + row0) * D_);
#pragma unroll
  for (int i = 0; i < 32; ++i) {                // 32x1024 f32 -> bf16 LDS
    int i4 = tid + i * 256;
    float4 f = src[i4];
    int e = i4 * 4;
    Xt[e + 0] = f2bf(f.x); Xt[e + 1] = f2bf(f.y);
    Xt[e + 2] = f2bf(f.z); Xt[e + 3] = f2bf(f.w);
  }
  __syncthreads();

  const unsigned short* arow = Xt + (rt * 16 + (lane & 15)) * 1024;
  const int col  = nt * 16 + (lane & 15);
  const int rsel = (lane >> 4) << 3;            // +8 rows for upper half-wave
  // W-slab transpose indices: idx = tid*8+j  ->  k = tid>>3, n = (tid&7)*8+j
  const int wkl = tid >> 3;
  const int wn0 = (tid & 7) * 8;

  for (int h = 0; h < H_; ++h) {
    v8f acc = {0.f,0.f,0.f,0.f,0.f,0.f,0.f,0.f};
    for (int kk = 0; kk < D_; kk += 32) {
      const float4* w4 = (const float4*)(W + (size_t)h * D_ * DK_ + (size_t)kk * DK_) + tid * 2;
      float4 wa = w4[0], wb = w4[1];
      Wl[(wn0 + 0) * 32 + wkl] = f2bf(wa.x);
      Wl[(wn0 + 1) * 32 + wkl] = f2bf(wa.y);
      Wl[(wn0 + 2) * 32 + wkl] = f2bf(wa.z);
      Wl[(wn0 + 3) * 32 + wkl] = f2bf(wa.w);
      Wl[(wn0 + 4) * 32 + wkl] = f2bf(wb.x);
      Wl[(wn0 + 5) * 32 + wkl] = f2bf(wb.y);
      Wl[(wn0 + 6) * 32 + wkl] = f2bf(wb.z);
      Wl[(wn0 + 7) * 32 + wkl] = f2bf(wb.w);
      __syncthreads();
      v16bf a  = frag_a_bf16(arow, kk, lane);
      v16bf bb = frag_b_bf16(Wl + col * 32 + ((lane >> 4) & 1) * 16);
      acc = wmma_bf16(a, bb, acc);
      __syncthreads();
    }
    const int l0 = row0 + rt * 16 + rsel;
    if (mat == 2) {                             // V: transposed store, contiguous in l
      unsigned short* dst = VsT + ((size_t)(h * B_ + b) * DK_ + col) * L_ + l0;
#pragma unroll
      for (int j = 0; j < 8; ++j) dst[j] = f2bf(acc[j]);
    } else {                                    // Q/K: row-major, stride DK_ in l
      unsigned short* dst = (mat == 0 ? Qs : Ks)
                          + ((size_t)(h * B_ + b) * L_ + l0) * DK_ + col;
#pragma unroll
      for (int j = 0; j < 8; ++j) dst[j * DK_] = f2bf(acc[j] * scale);
    }
  }
}

// ---------------------------------------------------------------------------
// Kernel 2: fused  S = Q Kᵀ  -> mask -> softmax -> attn(out) -> O = P V
// One WG = (h, b, 64 query rows).  64x1024 f32 score slab in LDS (256 KB).
// ---------------------------------------------------------------------------
__global__ __launch_bounds__(256) void k_attn(
    const unsigned short* __restrict__ Qs, const unsigned short* __restrict__ Ks,
    const unsigned short* __restrict__ VsT, const unsigned char* __restrict__ mask,
    float* __restrict__ attn, unsigned short* __restrict__ Ocat)
{
  extern __shared__ char smem[];
  float* Sc = (float*)smem;                     // [64][1024] scores/probs
  __shared__ float red[64][4];
  __shared__ float rstat[64];

  const int h = blockIdx.z, b = blockIdx.y;
  const int q0 = blockIdx.x * 64;
  const int tid = threadIdx.x, lane = tid & 31, wave = tid >> 5;
  const size_t hb = (size_t)(h * B_ + b);

  // ---- phase 1: scores (Q already scaled by 1/32)
  {
    const int rg = wave >> 1, nh = wave & 1;    // 4 row-groups x 2 key-halves
    const unsigned short* qrow = Qs + (hb * L_ + q0 + rg * 16 + (lane & 15)) * DK_;
    v16bf a0 = frag_a_bf16(qrow, 0, lane);
    v16bf a1 = frag_a_bf16(qrow, 32, lane);
    for (int t = 0; t < 32; ++t) {
      int n0 = nh * 512 + t * 16;
      const unsigned short* kcol =
          Ks + (hb * L_ + n0 + (lane & 15)) * DK_ + ((lane >> 4) & 1) * 16;
      v8f acc = {0.f,0.f,0.f,0.f,0.f,0.f,0.f,0.f};
      acc = wmma_bf16(a0, frag_b_bf16(kcol), acc);
      acc = wmma_bf16(a1, frag_b_bf16(kcol + 32), acc);
      int col = n0 + (lane & 15);
#pragma unroll
      for (int j = 0; j < 8; ++j)
        Sc[(rg * 16 + j + ((lane >> 4) << 3)) * 1024 + col] = acc[j];
    }
  }
  __syncthreads();

  // ---- phase 2: mask + softmax, write attn probs to d_out
  {
    const int row = tid >> 2, part = tid & 3;
    const int qg = q0 + row;
    const unsigned char* mrow = mask + ((size_t)b * L_ + qg) * L_;
    float* srow = Sc + row * 1024;
    float mx = -__builtin_inff();
    for (int i = 0; i < 256; ++i) {
      int c = part + i * 4;
      float s = srow[c];
      if (mrow[c]) { s = -__builtin_inff(); srow[c] = s; }
      mx = fmaxf(mx, s);
    }
    red[row][part] = mx; __syncthreads();
    if (tid < 64)
      rstat[tid] = fmaxf(fmaxf(red[tid][0], red[tid][1]), fmaxf(red[tid][2], red[tid][3]));
    __syncthreads();
    float m = rstat[row];
    float sum = 0.f;
    for (int i = 0; i < 256; ++i) {
      int c = part + i * 4;
      float e = __expf(srow[c] - m);
      srow[c] = e; sum += e;
    }
    __syncthreads();
    red[row][part] = sum; __syncthreads();
    if (tid < 64)
      rstat[tid] = 1.0f / (red[tid][0] + red[tid][1] + red[tid][2] + red[tid][3]);
    __syncthreads();
    float rinv = rstat[row];
    float* aout = attn + (hb * L_ + qg) * L_;
    for (int i = 0; i < 256; ++i) {
      int c = part + i * 4;
      float p = srow[c] * rinv;
      srow[c] = p;
      aout[c] = p;
    }
  }
  __syncthreads();

  // ---- phase 3: O = P @ V   (P f32 in LDS -> bf16 frags; V pre-transposed)
  {
    const int t0 = wave * 2;
    const int rg = t0 >> 2;                     // shared by both tiles of this wave
    const int vt0 = t0 & 3, vt1 = (t0 + 1) & 3;
    const float* prow = Sc + (rg * 16 + (lane & 15)) * 1024;
    const unsigned short* vbase = VsT + hb * DK_ * L_;
    const unsigned short* vcol0 =
        vbase + (size_t)(vt0 * 16 + (lane & 15)) * L_ + ((lane >> 4) & 1) * 16;
    const unsigned short* vcol1 =
        vbase + (size_t)(vt1 * 16 + (lane & 15)) * L_ + ((lane >> 4) & 1) * 16;
    v8f acc0 = {0.f,0.f,0.f,0.f,0.f,0.f,0.f,0.f};
    v8f acc1 = {0.f,0.f,0.f,0.f,0.f,0.f,0.f,0.f};
    for (int kk = 0; kk < L_; kk += 32) {
      v16bf a = frag_a_f32(prow, kk, lane);
      acc0 = wmma_bf16(a, frag_b_bf16(vcol0 + kk), acc0);
      acc1 = wmma_bf16(a, frag_b_bf16(vcol1 + kk), acc1);
    }
#pragma unroll
    for (int j = 0; j < 8; ++j) {
      int qg = q0 + rg * 16 + j + ((lane >> 4) << 3);
      unsigned short* orow = Ocat + (size_t)(b * L_ + qg) * (H_ * DK_) + h * DK_;
      orow[vt0 * 16 + (lane & 15)] = f2bf(acc0[j]);
      orow[vt1 * 16 + (lane & 15)] = f2bf(acc1[j]);
    }
  }
}

// ---------------------------------------------------------------------------
// Kernel 3: out = LN( Ocat @ proj_wᵀ + bias + residual ).  One WG = 16 rows.
// Ocat and proj_w both pre-converted to bf16 -> pure b128 fragment loads.
// ---------------------------------------------------------------------------
__global__ __launch_bounds__(256) void k_proj_ln(
    const unsigned short* __restrict__ Ocat, const unsigned short* __restrict__ pwb,
    const float* __restrict__ pb, const float* __restrict__ resid,
    const float* __restrict__ gamma, const float* __restrict__ beta,
    float* __restrict__ outp)
{
  extern __shared__ char smem[];
  float* Xr = (float*)smem;                     // [16][1024] pre-LN rows
  __shared__ float psum[16][17], psq[16][17];
  __shared__ float rmean[16], rrstd[16];

  const int m0 = blockIdx.x * 16;
  const int tid = threadIdx.x, lane = tid & 31, wave = tid >> 5;
  const int n0w = wave * 128;                   // each wave: 8 N-tiles

  v8f acc[8];
#pragma unroll
  for (int i = 0; i < 8; ++i) acc[i] = (v8f){0.f,0.f,0.f,0.f,0.f,0.f,0.f,0.f};

  const unsigned short* arow = Ocat + (size_t)(m0 + (lane & 15)) * 1024;
  for (int kk = 0; kk < 1024; kk += 32) {
    v16bf a = frag_a_bf16(arow, kk, lane);
#pragma unroll
    for (int nt = 0; nt < 8; ++nt) {
      const unsigned short* bp = pwb + (size_t)(n0w + nt * 16 + (lane & 15)) * 1024
                                     + kk + ((lane >> 4) & 1) * 16;
      __builtin_prefetch(bp + 32, 0, 1);        // global_prefetch_b8 on weight stream
      acc[nt] = wmma_bf16(a, frag_b_bf16(bp), acc[nt]);
    }
  }
#pragma unroll
  for (int nt = 0; nt < 8; ++nt) {
    int n = n0w + nt * 16 + (lane & 15);
#pragma unroll
    for (int j = 0; j < 8; ++j) {
      int r = j + ((lane >> 4) << 3);
      Xr[r * 1024 + n] = acc[nt][j] + pb[n] + resid[(size_t)(m0 + r) * 1024 + n];
    }
  }
  __syncthreads();

  const int r = tid >> 4, s = tid & 15;
  float sm = 0.f, sq = 0.f;
  for (int i = 0; i < 64; ++i) {
    float x = Xr[r * 1024 + s + i * 16];
    sm += x; sq += x * x;
  }
  psum[r][s] = sm; psq[r][s] = sq;
  __syncthreads();
  if (tid < 16) {
    float S = 0.f, Q = 0.f;
    for (int i = 0; i < 16; ++i) { S += psum[tid][i]; Q += psq[tid][i]; }
    float mean = S * (1.0f / 1024.0f);
    float var  = Q * (1.0f / 1024.0f) - mean * mean;
    rmean[tid] = mean;
    rrstd[tid] = rsqrtf(var + 1e-5f);
  }
  __syncthreads();
  float mean = rmean[r], rs = rrstd[r];
  float* orow = outp + (size_t)(m0 + r) * 1024;
  for (int i = 0; i < 64; ++i) {
    int c = s + i * 16;
    orow[c] = (Xr[r * 1024 + c] - mean) * rs * gamma[c] + beta[c];
  }
}

// ---------------------------------------------------------------------------
extern "C" void kernel_launch(void* const* d_in, const int* in_sizes, int n_in,
                              void* d_out, int out_size, void* d_ws, size_t ws_size,
                              hipStream_t stream) {
  (void)in_sizes; (void)n_in; (void)out_size; (void)ws_size;
  const float*         q    = (const float*)d_in[0];
  const float*         k    = (const float*)d_in[1];
  const float*         v    = (const float*)d_in[2];
  const unsigned char* mask = (const unsigned char*)d_in[3];
  const float*         wq   = (const float*)d_in[4];
  const float*         wk   = (const float*)d_in[5];
  const float*         wv   = (const float*)d_in[6];
  const float*         pw   = (const float*)d_in[7];
  const float*         pb   = (const float*)d_in[8];
  const float*         g    = (const float*)d_in[9];
  const float*         be   = (const float*)d_in[10];

  // scratch (all bf16): Qs/Ks/VsT (3x16 MB) + Ocat (16 MB) + proj_w (2 MB) = 66 MB
  unsigned short* Qs   = (unsigned short*)d_ws;
  unsigned short* Ks   = Qs  + (size_t)H_ * B_ * L_ * DK_;
  unsigned short* VsT  = Ks  + (size_t)H_ * B_ * L_ * DK_;
  unsigned short* Ocat = VsT + (size_t)H_ * B_ * L_ * DK_;
  unsigned short* PWb  = Ocat + (size_t)B_ * L_ * (H_ * DK_);

  float* outp = (float*)d_out;
  float* attn = outp + (size_t)B_ * L_ * D_;

  k_convw<<<dim3(D_ * D_ / 1024), 256, 0, stream>>>(pw, PWb);
  k_qkv_proj<<<dim3(96, B_), 256, 32 * 1024 * 2 + 64 * 32 * 2, stream>>>(
      q, k, v, wq, wk, wv, Qs, Ks, VsT);
  k_attn<<<dim3(L_ / 64, B_, H_), 256, 64 * 1024 * 4, stream>>>(
      Qs, Ks, VsT, mask, attn, Ocat);
  k_proj_ln<<<dim3(B_ * L_ / 16), 256, 16 * 1024 * 4, stream>>>(
      Ocat, PWb, pb, q, g, be, outp);
}
